// RNN_45569603010722
// MI455X (gfx1250) — matrix operand order repaired and checked by reading, not compiled, not run on hardware
//
#include <hip/hip_runtime.h>
#include <hip/hip_bf16.h>

// Problem dims (fixed by reference)
#define BB 128
#define SS 1024
#define II 512
#define HH 1024
#define OO 512
#define KK 1536            // I + H
#define NBLK 48            // persistent blocks (all resident; cheap device barrier)
#define THREADS 256        // 8 waves of 32
#define WPB 8

// K pipeline over LDS: 4 chunks of 384, double buffered.
#define NCHUNK 4
#define CK 384             // K elements per chunk
#define LDSTR 392          // padded LDS row stride (u16): 784B -> rows 4 banks apart
#define GRANPC 768         // 16-elem granules per chunk (32 rows * 24)

typedef __attribute__((ext_vector_type(16))) __bf16 v16bf;
typedef __attribute__((ext_vector_type(8)))  __bf16 v8bf;
typedef __attribute__((ext_vector_type(8)))  float  v8f;
typedef __attribute__((ext_vector_type(4)))  int    v4i;

union AFrag { v16bf v; v8bf h[2]; };

#define AS1 __attribute__((address_space(1)))
#define AS3 __attribute__((address_space(3)))

#if __has_builtin(__builtin_amdgcn_global_load_async_to_lds_b128)
#define HAVE_ASYNC_LDS 1
#endif

#if defined(HAVE_ASYNC_LDS)
# if __has_builtin(__builtin_amdgcn_s_wait_asynccnt)
#  define WAIT_ASYNC(n) __builtin_amdgcn_s_wait_asynccnt(n)
# else
#  define WAIT_ASYNC(n) asm volatile("s_wait_asynccnt " #n ::: "memory")
# endif
#else
# define WAIT_ASYNC(n) do {} while (0)
#endif

__device__ __forceinline__ unsigned short f32_to_bf16(float f) {
  unsigned u = __builtin_bit_cast(unsigned, f);
  u += 0x7FFFu + ((u >> 16) & 1u);          // round-to-nearest-even
  return (unsigned short)(u >> 16);
}
__device__ __forceinline__ float bf16_to_f32(unsigned short s) {
  return __builtin_bit_cast(float, ((unsigned)s) << 16);
}

// One-time pack of [W_ih ; W_io] (f32, row-major over K=1536) into bf16.
// 4.5 MB -> stays hot in the 192 MB L2 for all 1024 steps.
__global__ void __launch_bounds__(THREADS) prep_weights(
    const float* __restrict__ wih, const float* __restrict__ wio,
    unsigned short* __restrict__ wall) {
  size_t idx = (size_t)blockIdx.x * THREADS + threadIdx.x;
  const size_t total = (size_t)(HH + OO) * KK;
  if (idx < total) {
    int n = (int)(idx / KK);
    int k = (int)(idx - (size_t)n * KK);
    float v = (n < HH) ? wih[(size_t)n * KK + k]
                       : wio[(size_t)(n - HH) * KK + k];
    wall[idx] = f32_to_bf16(v);
  }
}

// Device-wide sense-free barrier: monotonically increasing generation counter.
__device__ __forceinline__ void grid_barrier(unsigned* cnt, unsigned phase) {
  __syncthreads();
  if (threadIdx.x == 0) {
    __threadfence();  // release all global writes (agent scope)
    __hip_atomic_fetch_add(cnt, 1u, __ATOMIC_RELEASE, __HIP_MEMORY_SCOPE_AGENT);
    const unsigned target = phase * (unsigned)NBLK;
    while (__hip_atomic_load(cnt, __ATOMIC_ACQUIRE, __HIP_MEMORY_SCOPE_AGENT) < target)
      __builtin_amdgcn_s_sleep(1);
    __threadfence();  // acquire
  }
  __syncthreads();
}

// Stage one K-chunk (32 rows x 384 K of bf16 = 24 KB) of the combined
// activation tile into LDS via GLOBAL_LOAD_ASYNC_TO_LDS_B128 (ASYNCcnt).
// Each thread issues exactly 3 x b128 -> per-wave ASYNCcnt += 3 per chunk.
__device__ __forceinline__ void stage_chunk(const unsigned short* __restrict__ cur,
                                            int r0, int kc0,
                                            unsigned short* __restrict__ dstLds,
                                            int tid) {
#pragma unroll
  for (int j = 0; j < 3; ++j) {
    int g   = tid + j * THREADS;        // granule id 0..767
    int row = g / 24;                   // 24 granules of 16 elems per row
    int gc  = g - row * 24;
    const unsigned short* gp = cur + (size_t)(r0 + row) * KK + kc0 + gc * 16;
    unsigned short*       lp = dstLds + row * LDSTR + gc * 16;
#if defined(HAVE_ASYNC_LDS)
    __builtin_amdgcn_global_load_async_to_lds_b128(
        (AS1 v4i*)gp, (AS3 v4i*)lp, 0, 0);
#else
    *(v8bf*)lp = *(const v8bf*)gp;      // synchronous fallback (still via LDS)
#endif
  }
}

// Persistent RNN kernel.
// Block = 1 row-group (32 batch rows) x 8 column tiles (its 8 waves).
// 4 row-groups x 12 col-groups = 48 blocks; 96 col tiles total:
// tiles 0..63 -> new hidden (tanh, bf16 into next combined buffer),
// tiles 64..95 -> out (f32 to d_out).
__global__ void __launch_bounds__(THREADS, 1) rnn_persist(
    const float* __restrict__ x,      // [B,S,I] f32
    const float* __restrict__ h0,     // [B,H]   f32
    const float* __restrict__ b_ih,   // [H]
    const float* __restrict__ b_io,   // [O]
    const unsigned short* __restrict__ wall,  // [(H+O), K] bf16
    unsigned short* __restrict__ buf0,        // [B, K] bf16 combined (ping)
    unsigned short* __restrict__ buf1,        // [B, K] bf16 combined (pong)
    float* __restrict__ out,          // [B,S,O] f32
    float* __restrict__ hfinal,       // [B,H]   f32
    unsigned* __restrict__ cnt)
{
  __shared__ unsigned short ldsA[2][32 * LDSTR];   // 2 x 24.5 KB, padded rows

  const int tid  = threadIdx.x;
  const int gtid = blockIdx.x * THREADS + tid;
  const int lane = tid & 31;
  const int nl   = lane & 15;
  const int hi   = lane >> 4;

  const int rowGroup = blockIdx.x & 3;           // 32-row A tile of this block
  const int colGroup = blockIdx.x >> 2;          // 0..11
  const int colTile  = colGroup * 8 + (tid >> 5);// 0..95
  const int r0       = rowGroup * 32;
  const int n        = colTile * 16 + nl;        // global output column 0..1535
  const bool isHidden = (colTile < 64);
  const float bias = isHidden ? b_ih[n] : b_io[n - HH];
  // B-matrix (32x16): lane holds weight row n, contiguous K = 16*hi..16*hi+15
  const unsigned short* wrow = wall + (size_t)n * KK + 16 * hi;

  // ---- initial fill of buf0 = [x_0 | h_0] (bf16) ----
  for (int idx = gtid; idx < BB * II; idx += NBLK * THREADS) {
    int b = idx / II, k = idx - b * II;
    buf0[b * KK + k] = f32_to_bf16(x[(size_t)b * SS * II + k]);  // t = 0
  }
  for (int idx = gtid; idx < BB * HH; idx += NBLK * THREADS) {
    int b = idx / HH, h = idx - b * HH;
    buf0[b * KK + II + h] = f32_to_bf16(h0[idx]);
  }
  grid_barrier(cnt, 1u);

  for (int t = 0; t < SS; ++t) {
    const unsigned short* cur = (t & 1) ? buf1 : buf0;
    unsigned short*       nxt = (t & 1) ? buf0 : buf1;

    v8f acc0 = {};
    v8f acc1 = {};

    // Software-pipelined: async-stage chunk c+1 while computing chunk c.
    stage_chunk(cur, r0, 0, &ldsA[0][0], tid);
    for (int c = 0; c < NCHUNK; ++c) {
      if (c + 1 < NCHUNK) {
        stage_chunk(cur, r0, (c + 1) * CK, &ldsA[(c + 1) & 1][0], tid);
        WAIT_ASYNC(3);            // chunk c's 3 b128s done (in-order), c+1 in flight
      } else {
        WAIT_ASYNC(0);
      }
      __syncthreads();            // chunk c visible block-wide

      const unsigned short* bufp = &ldsA[c & 1][0];
      // A (16x32 bf16): lane row = l&15,
      // contiguous K {16hi..16hi+15} (matches B) -> two 16B ds loads.
      const unsigned short* pa = bufp + nl * LDSTR + 16 * hi;
#pragma unroll 4
      for (int kl = 0; kl < CK; kl += 32) {
        AFrag a0, a1;
        v16bf bf;
        a0.h[0] = *(const v8bf*)(pa + kl);
        a0.h[1] = *(const v8bf*)(pa + kl + 8);
        a1.h[0] = *(const v8bf*)(pa + kl + 16 * LDSTR);
        a1.h[1] = *(const v8bf*)(pa + kl + 16 * LDSTR + 8);
        bf      = *(const v16bf*)(wrow + c * CK + kl);
        acc0 = __builtin_amdgcn_wmma_f32_16x16x32_bf16(
            false, a0.v, false, bf, (short)0, acc0, false, false);
        acc1 = __builtin_amdgcn_wmma_f32_16x16x32_bf16(
            false, a1.v, false, bf, (short)0, acc1, false, false);
      }
      __syncthreads();            // all waves done with ldsA[c&1] before reuse
    }

    // ---- epilogue: C/D layout -> element i is (row 8*hi+i, col nl) ----
    if (isHidden) {
      const int col = colTile * 16 + nl;   // 0..1023
#pragma unroll
      for (int i = 0; i < 8; ++i) {
        const int b0 = r0 + 8 * hi + i;
        const int b1 = b0 + 16;
        nxt[(size_t)b0 * KK + II + col] = f32_to_bf16(tanhf(acc0[i] + bias));
        nxt[(size_t)b1 * KK + II + col] = f32_to_bf16(tanhf(acc1[i] + bias));
      }
    } else {
      const int col = (colTile - 64) * 16 + nl;   // 0..511
#pragma unroll
      for (int i = 0; i < 8; ++i) {
        const int b0 = r0 + 8 * hi + i;
        const int b1 = b0 + 16;
        out[(size_t)b0 * SS * OO + (size_t)t * OO + col] = acc0[i] + bias;
        out[(size_t)b1 * SS * OO + (size_t)t * OO + col] = acc1[i] + bias;
      }
    }

    // ---- cooperative f32->bf16 stage of x_{t+1} into nxt (disjoint region) ----
    if (t + 1 < SS) {
      for (int idx = gtid; idx < BB * II; idx += NBLK * THREADS) {
        int b = idx / II, k = idx - b * II;
        nxt[b * KK + k] =
            f32_to_bf16(x[(size_t)b * SS * II + (size_t)(t + 1) * II + k]);
      }
    }
    grid_barrier(cnt, (unsigned)(t + 2));
  }

  // S is even -> final hidden lives in buf0's hidden region.
  for (int idx = gtid; idx < BB * HH; idx += NBLK * THREADS) {
    int b = idx / HH, h = idx - b * HH;
    hfinal[idx] = bf16_to_f32(buf0[b * KK + II + h]);
  }
}

extern "C" void kernel_launch(void* const* d_in, const int* in_sizes, int n_in,
                              void* d_out, int out_size, void* d_ws, size_t ws_size,
                              hipStream_t stream) {
  const float* x    = (const float*)d_in[0];
  const float* h0   = (const float*)d_in[1];
  const float* Wih  = (const float*)d_in[2];
  const float* bih  = (const float*)d_in[3];
  const float* Wio  = (const float*)d_in[4];
  const float* bio  = (const float*)d_in[5];

  float* outputs = (float*)d_out;                          // [B,S,O]
  float* hfinal  = outputs + (size_t)BB * SS * OO;         // [B,H]

  // Workspace layout (all 16B-aligned): wall | buf0 | buf1 | counter
  unsigned short* wall = (unsigned short*)d_ws;            // (H+O)*K bf16
  unsigned short* buf0 = wall + (size_t)(HH + OO) * KK;    // B*K bf16
  unsigned short* buf1 = buf0 + (size_t)BB * KK;
  unsigned* cnt = (unsigned*)(buf1 + (size_t)BB * KK);
  const size_t need = (size_t)(HH + OO) * KK * 2 + 2 * (size_t)BB * KK * 2 + 64;
  if (ws_size < need) return;  // deterministic guard (ws_size fixed per harness)

  (void)hipMemsetAsync(cnt, 0, sizeof(unsigned), stream);

  const int wtotal = (HH + OO) * KK;
  prep_weights<<<(wtotal + THREADS - 1) / THREADS, THREADS, 0, stream>>>(
      Wih, Wio, wall);

  rnn_persist<<<NBLK, THREADS, 0, stream>>>(
      x, h0, bih, bio, wall, buf0, buf1, outputs, hfinal, cnt);
}
